// MoELayer_55722905699096
// MI455X (gfx1250) — matrix operand order repaired and checked by reading, not compiled, not run on hardware
//
#include <hip/hip_runtime.h>
#include <hip/hip_bf16.h>

// ---------------------------------------------------------------------------
// MoE forward for MI455X (gfx1250, wave32, WMMA bf16 16x16x32).
// x:[4,2048,1024] f32, gate_w:[1024,16], Wg/Wu:[16,1024,2816], Wd:[16,2816,1024]
// out: [4,2048,1024] f32  ++  aux_loss scalar (concatenated in d_out)
// ---------------------------------------------------------------------------

typedef __attribute__((ext_vector_type(16))) __bf16 v16bf;
typedef __attribute__((ext_vector_type(8)))  float  v8f;

#define T_TOK   8192
#define D_DIM   1024
#define I_DIM   2816
#define E_NUM   16
#define TM      32          // token rows per block
#define ICHUNK  256         // I processed per outer chunk (8 waves * 32)
#define XLD     (D_DIM + 8) // padded LDS stride (bank rotation)
#define HLD     (ICHUNK + 8)

__device__ __forceinline__ v8f wmma_bf16(v16bf a, v16bf b, v8f c) {
  return __builtin_amdgcn_wmma_f32_16x16x32_bf16(
      /*neg_a=*/false, a, /*neg_b=*/false, b,
      /*c_mod=*/(short)0, c, /*reuse_a=*/false, /*reuse_b=*/false);
}

// A fragment (16x32, bf16) from row-major bf16 LDS tile.
// lane<16: row=r0+lane, elem e = K k0+e
// lane>=16: row=r0+lane-16, elem e = K k0+16+e
__device__ __forceinline__ v16bf load_A(const __bf16* base, int ld, int r0,
                                        int k0, int lane) {
  const __bf16* p = base + (size_t)(r0 + (lane & 15)) * ld + k0 + ((lane >> 4) << 4);
  v16bf a;
#pragma unroll
  for (int i = 0; i < 8; ++i) { a[i] = p[i]; a[i + 8] = p[i + 8]; }
  return a;
}

// B fragment (32x16, bf16) from row-major f32 weights W[K][N] with leading dim ld.
// lane<16: N=n0+lane, elem e = K k0+e ; lane>=16: N=n0+lane-16, elem e = K k0+16+e
__device__ __forceinline__ v16bf load_B(const float* W, int ld, int k0, int n0,
                                        int lane) {
  const float* p = W + (size_t)(k0 + ((lane >> 4) << 4)) * ld + n0 + (lane & 15);
  v16bf b;
#pragma unroll
  for (int e = 0; e < 16; ++e) b[e] = (__bf16)p[(size_t)e * ld];
  return b;
}

// Store a 16x16 f32 C/D fragment as bf16 into row-major LDS tile.
// elem r: row = r0 + (lane>>4)*8 + r, col = c0 + (lane&15)
__device__ __forceinline__ void store_C_bf16(__bf16* base, int ld, int r0,
                                             int c0, int lane, v8f c) {
  int row0 = r0 + ((lane >> 4) << 3);
  int col  = c0 + (lane & 15);
#pragma unroll
  for (int r = 0; r < 8; ++r) base[(size_t)(row0 + r) * ld + col] = (__bf16)c[r];
}

// ---------------------------------------------------------------------------
__global__ void zero_out_kernel(float* p, int n4) {
  int i = blockIdx.x * blockDim.x + threadIdx.x;
  if (i < n4) ((float4*)p)[i] = make_float4(0.f, 0.f, 0.f, 0.f);
}

__global__ void zero_meta_kernel(int* counts, float* probSum, float* maskCnt) {
  int i = threadIdx.x;
  if (i < E_NUM) { counts[i] = 0; probSum[i] = 0.f; maskCnt[i] = 0.f; }
}

// ---------------------------------------------------------------------------
// Router: one wave per token. logits -> softmax -> top2 -> renormalized
// combine weights; build per-expert token lists; aux-loss statistics.
__global__ __launch_bounds__(256) void moe_router_kernel(
    const float* __restrict__ x, const float* __restrict__ gate_w,
    float* __restrict__ combine_w, int* __restrict__ lists,
    int* __restrict__ counts, float* __restrict__ probSum,
    float* __restrict__ maskCnt) {
  int lane = threadIdx.x & 31;
  int t = blockIdx.x * 8 + (threadIdx.x >> 5);

  float acc[E_NUM];
#pragma unroll
  for (int e = 0; e < E_NUM; ++e) acc[e] = 0.f;

  const float* xr = x + (size_t)t * D_DIM;
  for (int d = lane; d < D_DIM; d += 32) {
    float xv = xr[d];
    const float* gw = gate_w + (size_t)d * E_NUM;
#pragma unroll
    for (int e = 0; e < E_NUM; ++e) acc[e] = fmaf(xv, gw[e], acc[e]);
  }
#pragma unroll
  for (int e = 0; e < E_NUM; ++e) {
#pragma unroll
    for (int off = 16; off > 0; off >>= 1)
      acc[e] += __shfl_xor(acc[e], off, 32);
  }
  // softmax (all lanes redundantly)
  float m = acc[0];
#pragma unroll
  for (int e = 1; e < E_NUM; ++e) m = fmaxf(m, acc[e]);
  float p[E_NUM], sum = 0.f;
#pragma unroll
  for (int e = 0; e < E_NUM; ++e) { p[e] = __expf(acc[e] - m); sum += p[e]; }
  float inv = 1.f / sum;
#pragma unroll
  for (int e = 0; e < E_NUM; ++e) p[e] *= inv;
  // top-2 (first occurrence on ties, matching lax.top_k)
  int i1 = 0; float p1 = p[0];
#pragma unroll
  for (int e = 1; e < E_NUM; ++e) if (p[e] > p1) { p1 = p[e]; i1 = e; }
  int i2 = (i1 == 0) ? 1 : 0; float p2 = -1.f; i2 = -1;
#pragma unroll
  for (int e = 0; e < E_NUM; ++e)
    if (e != i1 && p[e] > p2) { p2 = p[e]; i2 = e; }
  float wn = 1.f / (p1 + p2);
  float w1 = p1 * wn, w2 = p2 * wn;

  if (lane < E_NUM) {
    float cw = (lane == i1) ? w1 : ((lane == i2) ? w2 : 0.f);
    combine_w[(size_t)t * E_NUM + lane] = cw;
    float myp = 0.f;
#pragma unroll
    for (int e = 0; e < E_NUM; ++e) myp = (lane == e) ? p[e] : myp;
    atomicAdd(probSum + lane, myp);
  }
  if (lane == 0) {
    atomicAdd(maskCnt + i1, 1.f);
    atomicAdd(maskCnt + i2, 1.f);
    int pos1 = atomicAdd(counts + i1, 1); lists[i1 * T_TOK + pos1] = t;
    int pos2 = atomicAdd(counts + i2, 1); lists[i2 * T_TOK + pos2] = t;
  }
}

__global__ void moe_aux_kernel(const float* __restrict__ probSum,
                               const float* __restrict__ maskCnt,
                               float* __restrict__ out_aux) {
  const float invT = 1.0f / (float)T_TOK;
  float s = 0.f;
#pragma unroll
  for (int e = 0; e < E_NUM; ++e) s += (maskCnt[e] * invT) * (probSum[e] * invT);
  *out_aux = 0.01f * s * (float)E_NUM;
}

// ---------------------------------------------------------------------------
// Fused expert kernel. Block = (token-tile of 32, expert). 8 waves.
// Y[32,1024] kept in VGPR accumulators across the I loop; per 256-wide I
// chunk each wave produces a 32-col slice of H = silu(X Wg) * (X Wu) into
// LDS (bf16), then all waves accumulate Y += H_chunk @ Wd_chunk.
__global__ __launch_bounds__(256, 1) void moe_expert_kernel(
    const float* __restrict__ x, const float* __restrict__ Wg,
    const float* __restrict__ Wu, const float* __restrict__ Wd,
    const float* __restrict__ combine_w, const int* __restrict__ lists,
    const int* __restrict__ counts, float* __restrict__ out) {
  extern __shared__ char smem[];
  __bf16* Xs     = (__bf16*)smem;                       // [TM][XLD]
  __bf16* Hs     = Xs + (size_t)TM * XLD;               // [TM][HLD]
  float*  wrow_s = (float*)(Hs + (size_t)TM * HLD);     // [TM]
  int*    tok_s  = (int*)(wrow_s + TM);                 // [TM]

  const int e    = blockIdx.y;
  const int tile = blockIdx.x;
  const int Ne   = counts[e];
  if (tile * TM >= Ne) return;  // uniform exit: EXEC stays all-ones for WMMA

  const int tid  = threadIdx.x;
  const int wave = tid >> 5;
  const int lane = tid & 31;

  if (tid < TM) {
    int row = tile * TM + tid;
    int tok = (row < Ne) ? lists[e * T_TOK + row] : lists[e * T_TOK];
    tok_s[tid]  = tok;
    wrow_s[tid] = (row < Ne) ? combine_w[(size_t)tok * E_NUM + e] : 0.f;
  }
  __syncthreads();

  // Stage X tile (fp32 -> bf16) into LDS.
  for (int i = tid; i < TM * (D_DIM / 4); i += 256) {
    int row = i / (D_DIM / 4);
    int c4  = i - row * (D_DIM / 4);
    float4 v = ((const float4*)(x + (size_t)tok_s[row] * D_DIM))[c4];
    __bf16* d = Xs + (size_t)row * XLD + c4 * 4;
    d[0] = (__bf16)v.x; d[1] = (__bf16)v.y; d[2] = (__bf16)v.z; d[3] = (__bf16)v.w;
  }
  __syncthreads();

  const float* wg = Wg + (size_t)e * D_DIM * I_DIM;
  const float* wu = Wu + (size_t)e * D_DIM * I_DIM;
  const float* wd = Wd + (size_t)e * I_DIM * D_DIM;

  v8f Yacc[2][8];
#pragma unroll
  for (int mt = 0; mt < 2; ++mt)
#pragma unroll
    for (int nt = 0; nt < 8; ++nt) Yacc[mt][nt] = (v8f)(0.f);

  const int n0 = wave * 128;  // this wave's 128-col slice of Y

  for (int ch = 0; ch < I_DIM; ch += ICHUNK) {
    // ---- H stage: this wave's 32 columns i0..i0+31 of the chunk ----
    const int i0 = ch + wave * 32;
    v8f g[2][2], u[2][2];
#pragma unroll
    for (int mt = 0; mt < 2; ++mt)
#pragma unroll
      for (int it = 0; it < 2; ++it) { g[mt][it] = (v8f)(0.f); u[mt][it] = (v8f)(0.f); }

    for (int k0 = 0; k0 < D_DIM; k0 += 32) {
      v16bf a0 = load_A(Xs, XLD, 0,  k0, lane);
      v16bf a1 = load_A(Xs, XLD, 16, k0, lane);
#pragma unroll
      for (int it = 0; it < 2; ++it) {
        v16bf bg = load_B(wg, I_DIM, k0, i0 + it * 16, lane);
        v16bf bu = load_B(wu, I_DIM, k0, i0 + it * 16, lane);
        g[0][it] = wmma_bf16(a0, bg, g[0][it]);
        g[1][it] = wmma_bf16(a1, bg, g[1][it]);
        u[0][it] = wmma_bf16(a0, bu, u[0][it]);
        u[1][it] = wmma_bf16(a1, bu, u[1][it]);
      }
    }

    __syncthreads();  // previous chunk's Hs fully consumed
#pragma unroll
    for (int mt = 0; mt < 2; ++mt)
#pragma unroll
      for (int it = 0; it < 2; ++it) {
        v8f hv;
#pragma unroll
        for (int r = 0; r < 8; ++r) {
          float gg = g[mt][it][r];
          float uu = u[mt][it][r];
          hv[r] = (gg * __frcp_rn(1.f + __expf(-gg))) * uu;  // silu(g)*u
        }
        store_C_bf16(Hs, HLD, mt * 16, wave * 32 + it * 16, lane, hv);
      }
    __syncthreads();  // Hs chunk ready for all waves

    // ---- Y stage: Yacc[32, n0:n0+128] += Hs[32, 0:256] @ Wd[ch:ch+256, ...]
#pragma unroll 1
    for (int kt = 0; kt < ICHUNK; kt += 32) {
      v16bf h0 = load_A(Hs, HLD, 0,  kt, lane);
      v16bf h1 = load_A(Hs, HLD, 16, kt, lane);
#pragma unroll
      for (int nt = 0; nt < 8; ++nt) {
        v16bf bd = load_B(wd, D_DIM, ch + kt, n0 + nt * 16, lane);
        Yacc[0][nt] = wmma_bf16(h0, bd, Yacc[0][nt]);
        Yacc[1][nt] = wmma_bf16(h1, bd, Yacc[1][nt]);
      }
    }
  }

  // ---- scatter: out[token, col] += Y * combine_weight ----
#pragma unroll
  for (int mt = 0; mt < 2; ++mt)
#pragma unroll
    for (int nt = 0; nt < 8; ++nt) {
      int row0 = mt * 16 + ((lane >> 4) << 3);
      int col  = n0 + nt * 16 + (lane & 15);
#pragma unroll
      for (int r = 0; r < 8; ++r) {
        int rr = row0 + r;
        float val = Yacc[mt][nt][r] * wrow_s[rr];
        atomicAdd(out + (size_t)tok_s[rr] * D_DIM + col, val);
      }
    }
}

// ---------------------------------------------------------------------------
extern "C" void kernel_launch(void* const* d_in, const int* in_sizes, int n_in,
                              void* d_out, int out_size, void* d_ws,
                              size_t ws_size, hipStream_t stream) {
  const float* x      = (const float*)d_in[0];
  const float* gate_w = (const float*)d_in[1];
  const float* Wg     = (const float*)d_in[2];
  const float* Wu     = (const float*)d_in[3];
  const float* Wd     = (const float*)d_in[4];
  float* out     = (float*)d_out;
  float* out_aux = out + (size_t)T_TOK * D_DIM;

  char* ws = (char*)d_ws;
  float* combine_w = (float*)ws;                                   // T*E f32
  int*   lists     = (int*)(ws + (size_t)T_TOK * E_NUM * 4);       // E*T i32
  char*  meta      = ws + (size_t)T_TOK * E_NUM * 8;
  int*   counts    = (int*)meta;                                   // E
  float* probSum   = (float*)(meta + 128);                         // E
  float* maskCnt   = (float*)(meta + 256);                         // E

  int n4 = T_TOK * D_DIM / 4;
  zero_out_kernel<<<(n4 + 255) / 256, 256, 0, stream>>>(out, n4);
  zero_meta_kernel<<<1, 64, 0, stream>>>(counts, probSum, maskCnt);
  moe_router_kernel<<<T_TOK / 8, 256, 0, stream>>>(x, gate_w, combine_w, lists,
                                                   counts, probSum, maskCnt);
  moe_aux_kernel<<<1, 1, 0, stream>>>(probSum, maskCnt, out_aux);

  dim3 grid(T_TOK / TM, E_NUM);
  size_t smem = (size_t)TM * XLD * sizeof(__bf16) +
                (size_t)TM * HLD * sizeof(__bf16) + TM * 4 + TM * 4;
  moe_expert_kernel<<<grid, 256, smem, stream>>>(x, Wg, Wu, Wd, combine_w,
                                                 lists, counts, out);
}
